// EfficientVSR_18897856103187
// MI455X (gfx1250) — matrix operand order repaired and checked by reading, not compiled, not run on hardware
//
#include <hip/hip_runtime.h>
#include <hip/hip_bf16.h>

typedef __attribute__((ext_vector_type(16))) __bf16 v16bf;
typedef __attribute__((ext_vector_type(8)))  __bf16 v8bf;
typedef __attribute__((ext_vector_type(8)))  float  v8f;

#define B_    8
#define C_    128
#define H_    256
#define W_    256
#define WS_   8
#define NH    4
#define HD    32
#define NTOK  64
#define NWIN  (B_ * (H_/WS_) * (W_/WS_))   // 8192

// LDS strides (in elements)
#define LDQ   136   // bf16 rows of 128 (+8 pad) -> 272B, 16B-aligned chunks
#define LDVT  72    // bf16 rows of 64 (+8 pad)  -> 144B
#define LDP   72    // bf16 rows of 64 (+8 pad)
#define LDS_S 68    // f32 rows of 64 (+4 pad)   -> 272B
#define LDO   132   // f32 rows of 128 (+4 pad)

// LDS byte offsets (manual aliasing: staging region reused for S and Outf)
#define OFF_STAGE_Q  0                       // qs : 64*136 bf16 = 17408B
#define OFF_STAGE_KV 17408                   // kvs: 17408B
#define OFF_SF       0                       // Sf : 64*68 f32 = 17408B (aliases qs, dead)
#define OFF_OUTF     0                       // Outf: 64*132 f32 = 33792B (aliases qs+kvs, dead)
#define OFF_QM       34816                   // Qm : 17408B
#define OFF_KM       52224                   // Km : 17408B
#define OFF_VT       69632                   // Vt : 128*72 bf16 = 18432B
#define OFF_PB       88064                   // Pb : 64*72 bf16 = 9216B
#define OFF_OB       97280                   // Ob : 17408B
#define OFF_STATS    114688                  // reduction scratch (floats, see below)
#define SMEM_BYTES   118272                  // 114688 + 896*4

// stats region layout (float indices)
#define ST_MU    0     // [64]
#define ST_RS    64    // [64]
#define ST_PMAX  128   // [256]
#define ST_PSUM  384   // [256]
#define ST_LNW   640   // [128]
#define ST_LNB   768   // [128]  -> 896 floats total

// workspace layout (bytes)
#define WS_WBF_BYTES  131072     // 4 * 128*128 bf16
#define WS_ROPE_OFF   131072     // 240 * float2

// ------------------- pre-pass: weights -> bf16, RoPE table ------------------
__global__ __launch_bounds__(256)
void vsr_prep_kernel(const float* __restrict__ Wq, const float* __restrict__ Wk,
                     const float* __restrict__ Wv, const float* __restrict__ Wo,
                     __bf16* __restrict__ wbf, float* __restrict__ ropetab)
{
    int gid = blockIdx.x * 256 + threadIdx.x;     // 0..16383 float4 groups
    const float* srcs[4] = {Wq, Wk, Wv, Wo};
    int m   = gid >> 12;                          // 4096 float4 per matrix
    int off = (gid & 4095) * 4;
    float4 v = *(const float4*)(srcs[m] + off);
    __bf16* dst = wbf + m * 16384 + off;
    dst[0] = (__bf16)v.x; dst[1] = (__bf16)v.y;
    dst[2] = (__bf16)v.z; dst[3] = (__bf16)v.w;

    if (blockIdx.x == 0 && threadIdx.x < 240) {   // 15 (y+x) values x 16 freqs
        int s = threadIdx.x >> 4;
        int i = threadIdx.x & 15;
        float ang = (float)s * __powf(10000.0f, -(float)i * (1.0f / 16.0f));
        ropetab[2 * threadIdx.x]     = __cosf(ang);
        ropetab[2 * threadIdx.x + 1] = __sinf(ang);
    }
}

// ------------------------------- main kernel --------------------------------
__global__ __launch_bounds__(256, 1)
void vsr_window_attn_kernel(const float* __restrict__ temporal,
                            const float* __restrict__ spatial,
                            const __bf16* __restrict__ wbf,
                            const float* __restrict__ ropetab,
                            const float* __restrict__ lnw,
                            const float* __restrict__ lnb,
                            float* __restrict__ out)
{
    extern __shared__ unsigned char smem[];
    __bf16* qs   = (__bf16*)(smem + OFF_STAGE_Q);
    __bf16* kvs  = (__bf16*)(smem + OFF_STAGE_KV);
    float*  Sf   = (float *)(smem + OFF_SF);
    float*  Outf = (float *)(smem + OFF_OUTF);
    __bf16* Qm   = (__bf16*)(smem + OFF_QM);
    __bf16* Km   = (__bf16*)(smem + OFF_KM);
    __bf16* Vt   = (__bf16*)(smem + OFF_VT);
    __bf16* Pb   = (__bf16*)(smem + OFF_PB);
    __bf16* Ob   = (__bf16*)(smem + OFF_OB);
    float*  stat = (float *)(smem + OFF_STATS);

    const __bf16* Wqb = wbf;
    const __bf16* Wkb = wbf + 16384;
    const __bf16* Wvb = wbf + 32768;
    const __bf16* Wob = wbf + 49152;

    const int tid  = threadIdx.x;
    const int lane = tid & 31;
    const int wid  = tid >> 5;          // 8 wave32 per block
    const int lm   = lane & 15;
    const int lhi  = (lane >> 4) & 1;

    const int win = blockIdx.x;
    const int b   = win >> 10;
    const int wh  = (win >> 5) & 31;
    const int ww  = win & 31;
    const int h0  = wh * WS_;
    const int w0  = ww * WS_;

    // CDNA5 global prefetch of bf16 weights into L2
    __builtin_prefetch((const void*)(wbf + tid * 256), 0, 3);

    // ---------------- Phase 0: stage windows to LDS bf16; LN params ---------
    if (tid < 128) stat[ST_LNW + tid] = lnw[tid];
    else           stat[ST_LNB + (tid - 128)] = lnb[tid - 128];

    for (int idx = tid; idx < NTOK * C_; idx += 256) {
        int c = idx >> 6;
        int n = idx & 63;
        int gh = h0 + (n >> 3);
        int gw = w0 + (n & 7);
        size_t g = (((size_t)(b * C_ + c)) * H_ + gh) * W_ + gw;
        qs [n * LDQ + c] = (__bf16)temporal[g];
        kvs[n * LDQ + c] = (__bf16)spatial[g];
    }
    __syncthreads();

    // --------- fragment loaders (ISA 16-bit A/B VGPR layouts, wave32) -------
    auto loadA = [&](const __bf16* base, int ld, int m0, int k0) -> v16bf {
        const __bf16* p = base + (size_t)(m0 + lm) * ld + k0 + lhi * 8;
        v8bf lo = *(const v8bf*)p;
        v8bf hi = *(const v8bf*)(p + 16);
        v16bf r;
        #pragma unroll
        for (int i = 0; i < 8; ++i) { r[i] = lo[i]; r[i + 8] = hi[i]; }
        return r;
    };
    auto loadB16 = [&](const __bf16* colptr) -> v16bf {   // K 8h..8h+7, 16+8h..16+8h+7
        const __bf16* p = colptr + lhi * 8;
        v8bf a = *(const v8bf*)p;
        v8bf c = *(const v8bf*)(p + 16);
        v16bf r;
        #pragma unroll
        for (int i = 0; i < 8; ++i) { r[i] = a[i]; r[i + 8] = c[i]; }
        return r;
    };
    auto loadB_W = [&](const __bf16* Wmat, int n0, int k0) -> v16bf {
        return loadB16(Wmat + (size_t)(n0 + lm) * C_ + k0);
    };

    // ---------------- Phase 1: Q/K/V projections ----------------------------
    for (int t = wid; t < 32; t += 8) {
        int mt = t >> 3;
        int nt = t & 7;
        v8f aq = {}, ak = {}, av = {};
        #pragma unroll
        for (int kt = 0; kt < 4; ++kt) {
            v16bf aQ  = loadA(qs,  LDQ, mt * 16, kt * 32);
            v16bf aKV = loadA(kvs, LDQ, mt * 16, kt * 32);
            v16bf bq  = loadB_W(Wqb, nt * 16, kt * 32);
            v16bf bk  = loadB_W(Wkb, nt * 16, kt * 32);
            v16bf bv  = loadB_W(Wvb, nt * 16, kt * 32);
            aq = __builtin_amdgcn_wmma_f32_16x16x32_bf16(false, aQ,  false, bq, (short)0, aq, false, false);
            ak = __builtin_amdgcn_wmma_f32_16x16x32_bf16(false, aKV, false, bk, (short)0, ak, false, false);
            av = __builtin_amdgcn_wmma_f32_16x16x32_bf16(false, aKV, false, bv, (short)0, av, false, false);
        }
        int n  = nt * 16 + lm;
        int mb = mt * 16 + lhi * 8;
        #pragma unroll
        for (int j = 0; j < 8; ++j) {
            Qm[(mb + j) * LDQ + n] = (__bf16)aq[j];
            Km[(mb + j) * LDQ + n] = (__bf16)ak[j];
            Vt[n * LDVT + (mb + j)] = (__bf16)av[j];   // V transposed
        }
    }
    __syncthreads();

    // ---------------- Phase 1b: RoPE on Q and K (table-driven) --------------
    for (int it = tid; it < 64 * 64; it += 256) {
        int n = it >> 6;
        int j = it & 63;
        int head = j >> 4;
        int i    = j & 15;
        int s    = (n >> 3) + (n & 7);
        float2 t2 = *(const float2*)(ropetab + 2 * (s * 16 + i));
        float cs = t2.x, sn = t2.y;
        int c1 = head * HD + i;
        int c2 = c1 + 16;
        float q1 = (float)Qm[n * LDQ + c1], q2 = (float)Qm[n * LDQ + c2];
        Qm[n * LDQ + c1] = (__bf16)(q1 * cs - q2 * sn);
        Qm[n * LDQ + c2] = (__bf16)(q1 * sn + q2 * cs);
        float k1 = (float)Km[n * LDQ + c1], k2 = (float)Km[n * LDQ + c2];
        Km[n * LDQ + c1] = (__bf16)(k1 * cs - k2 * sn);
        Km[n * LDQ + c2] = (__bf16)(k1 * sn + k2 * cs);
    }
    __syncthreads();

    // ---------------- Phase 2: attention, one head at a time ----------------
    const float scale = 0.17677669529663687f;   // 1/sqrt(32)
    const int r4 = tid >> 2;                    // row for 4-thread reductions
    const int q4 = tid & 3;                     // quarter
    for (int head = 0; head < NH; ++head) {
        // scores S = Qh(64x32) * Kh^T(32x64)
        for (int t = wid; t < 16; t += 8) {
            int mt = t >> 2, nt = t & 3;
            v16bf a  = loadA(Qm, LDQ, mt * 16, head * HD);
            v16bf bm = loadB16(Km + (size_t)(nt * 16 + lm) * LDQ + head * HD);
            v8f acc = {};
            acc = __builtin_amdgcn_wmma_f32_16x16x32_bf16(false, a, false, bm, (short)0, acc, false, false);
            int n  = nt * 16 + lm;
            int mb = mt * 16 + lhi * 8;
            #pragma unroll
            for (int j = 0; j < 8; ++j) Sf[(mb + j) * LDS_S + n] = acc[j];
        }
        __syncthreads();

        // softmax: 4 threads per row
        {
            const float* row = Sf + r4 * LDS_S + q4 * 16;
            float mx = row[0];
            #pragma unroll
            for (int c = 1; c < 16; ++c) mx = fmaxf(mx, row[c]);
            stat[ST_PMAX + tid] = mx;
        }
        __syncthreads();
        {
            float mx = fmaxf(fmaxf(stat[ST_PMAX + r4 * 4 + 0], stat[ST_PMAX + r4 * 4 + 1]),
                             fmaxf(stat[ST_PMAX + r4 * 4 + 2], stat[ST_PMAX + r4 * 4 + 3]));
            float* row = Sf + r4 * LDS_S + q4 * 16;
            float s = 0.f;
            #pragma unroll
            for (int c = 0; c < 16; ++c) {
                float e = __expf((row[c] - mx) * scale);
                row[c] = e;
                s += e;
            }
            stat[ST_PSUM + tid] = s;
        }
        __syncthreads();
        {
            float inv = 1.0f / (stat[ST_PSUM + r4 * 4 + 0] + stat[ST_PSUM + r4 * 4 + 1] +
                                stat[ST_PSUM + r4 * 4 + 2] + stat[ST_PSUM + r4 * 4 + 3]);
            const float* row = Sf + r4 * LDS_S + q4 * 16;
            __bf16* prow = Pb + r4 * LDP + q4 * 16;
            #pragma unroll
            for (int c = 0; c < 16; ++c) prow[c] = (__bf16)(row[c] * inv);
        }
        __syncthreads();

        // O_h = P(64x64) * V_h(64x32) -- one 16x16 tile per wave
        {
            int mt = wid >> 1, nt = wid & 1;
            v8f acc = {};
            #pragma unroll
            for (int kt = 0; kt < 2; ++kt) {
                v16bf a  = loadA(Pb, LDP, mt * 16, kt * 32);
                v16bf bm = loadB16(Vt + (size_t)(head * HD + nt * 16 + lm) * LDVT + kt * 32);
                acc = __builtin_amdgcn_wmma_f32_16x16x32_bf16(false, a, false, bm, (short)0, acc, false, false);
            }
            int n  = head * HD + nt * 16 + lm;
            int mb = mt * 16 + lhi * 8;
            #pragma unroll
            for (int j = 0; j < 8; ++j) Ob[(mb + j) * LDQ + n] = (__bf16)acc[j];
        }
        __syncthreads();
    }

    // ---------------- Phase 3: output projection ----------------------------
    for (int t = wid; t < 32; t += 8) {
        int mt = t >> 3, nt = t & 7;
        v8f acc = {};
        #pragma unroll
        for (int kt = 0; kt < 4; ++kt) {
            v16bf a  = loadA(Ob, LDQ, mt * 16, kt * 32);
            v16bf bm = loadB_W(Wob, nt * 16, kt * 32);
            acc = __builtin_amdgcn_wmma_f32_16x16x32_bf16(false, a, false, bm, (short)0, acc, false, false);
        }
        int n  = nt * 16 + lm;
        int mb = mt * 16 + lhi * 8;
        #pragma unroll
        for (int j = 0; j < 8; ++j) Outf[(mb + j) * LDO + n] = acc[j];
    }
    __syncthreads();

    // ---------------- Phase 4: LayerNorm stats (4 threads per row) ----------
    {
        const float* row = Outf + r4 * LDO + q4 * 32;
        float s = 0.f, s2 = 0.f;
        #pragma unroll
        for (int c = 0; c < 32; ++c) { float v = row[c]; s += v; s2 += v * v; }
        stat[ST_PMAX + tid] = s;
        stat[ST_PSUM + tid] = s2;
    }
    __syncthreads();
    if (tid < 64) {
        float s  = stat[ST_PMAX + tid * 4 + 0] + stat[ST_PMAX + tid * 4 + 1] +
                   stat[ST_PMAX + tid * 4 + 2] + stat[ST_PMAX + tid * 4 + 3];
        float s2 = stat[ST_PSUM + tid * 4 + 0] + stat[ST_PSUM + tid * 4 + 1] +
                   stat[ST_PSUM + tid * 4 + 2] + stat[ST_PSUM + tid * 4 + 3];
        float mu  = s * (1.0f / C_);
        float var = s2 * (1.0f / C_) - mu * mu;
        stat[ST_MU + tid] = mu;
        stat[ST_RS + tid] = rsqrtf(var + 1e-5f);
    }
    __syncthreads();

    // ---------------- Phase 5: LN apply + residual + scatter ----------------
    for (int idx = tid; idx < NTOK * C_; idx += 256) {
        int c = idx >> 6;
        int n = idx & 63;
        int gh = h0 + (n >> 3);
        int gw = w0 + (n & 7);
        size_t g = (((size_t)(b * C_ + c)) * H_ + gh) * W_ + gw;
        float v = (Outf[n * LDO + c] - stat[ST_MU + n]) * stat[ST_RS + n] *
                  stat[ST_LNW + c] + stat[ST_LNB + c];
        out[g] = v + temporal[g];
    }
}

extern "C" void kernel_launch(void* const* d_in, const int* in_sizes, int n_in,
                              void* d_out, int out_size, void* d_ws, size_t ws_size,
                              hipStream_t stream) {
    (void)in_sizes; (void)n_in; (void)out_size; (void)ws_size;
    const float* temporal = (const float*)d_in[0];
    const float* spatial  = (const float*)d_in[1];
    const float* Wq       = (const float*)d_in[2];
    const float* Wk       = (const float*)d_in[3];
    const float* Wv       = (const float*)d_in[4];
    const float* Wo       = (const float*)d_in[5];
    const float* lnw      = (const float*)d_in[6];
    const float* lnb      = (const float*)d_in[7];
    float* out            = (float*)d_out;

    __bf16* wbf     = (__bf16*)d_ws;
    float*  ropetab = (float*)((char*)d_ws + WS_ROPE_OFF);

    hipFuncSetAttribute((const void*)vsr_window_attn_kernel,
                        hipFuncAttributeMaxDynamicSharedMemorySize, SMEM_BYTES);

    hipLaunchKernelGGL(vsr_prep_kernel, dim3(64), dim3(256), 0, stream,
                       Wq, Wk, Wv, Wo, wbf, ropetab);
    hipLaunchKernelGGL(vsr_window_attn_kernel, dim3(NWIN), dim3(256), SMEM_BYTES, stream,
                       temporal, spatial, wbf, ropetab, lnw, lnb, out);
}
